// Diffuser_18373870092389
// MI455X (gfx1250) — matrix operand order repaired and checked by reading, not compiled, not run on hardware
//
#include <hip/hip_runtime.h>
#include <hip/hip_bf16.h>
#include <stdint.h>

typedef __bf16 bf16;
typedef __attribute__((ext_vector_type(16))) __bf16 v16bf;
typedef __attribute__((ext_vector_type(8)))  float  v8f;
typedef __attribute__((ext_vector_type(4)))  int    v4i;

#define S_   1024
#define CA_  768
#define CS_  384
#define CZ_  64
#define H_   16
#define L_   2
#define C_   48
#define NCA_ 1536   /* N*CA */
#define KVG_ 2304   /* 3*CA */

/* ---- CDNA5 async global->LDS (guarded; falls back to sync staging) ---- */
#if defined(__has_builtin)
# if __has_builtin(__builtin_amdgcn_global_load_async_to_lds_b128)
#  define HAVE_ASYNC 1
# endif
# if __has_builtin(__builtin_amdgcn_s_wait_asynccnt)
#  define HAVE_WAIT_ASYNC_BUILTIN 1
# endif
#endif
#ifndef HAVE_ASYNC
# define HAVE_ASYNC 0
#endif
#ifndef HAVE_WAIT_ASYNC_BUILTIN
# define HAVE_WAIT_ASYNC_BUILTIN 0
#endif

__device__ __forceinline__ void async_copy16(const bf16* g, bf16* l) {
#if HAVE_ASYNC
  __builtin_amdgcn_global_load_async_to_lds_b128(
      (__attribute__((address_space(1))) v4i*)(void*)g,
      (__attribute__((address_space(3))) v4i*)(void*)l, 0, 0);
#else
  (void)g; (void)l;
#endif
}
__device__ __forceinline__ void wait_async() {
#if HAVE_ASYNC
# if HAVE_WAIT_ASYNC_BUILTIN
  __builtin_amdgcn_s_wait_asynccnt(0);
# else
  asm volatile("s_wait_asynccnt 0x0" ::: "memory");
# endif
#endif
}

/* ---------------- generic batched bf16 WMMA GEMM ----------------
 * C[M,N] = A[M,K] * B^T   (B stored [N][K] when bkmajor==0, [K][N] when 1)
 * epilogue modes: 0 = f32 (+bias), 1 = bf16 (+bias), 2 = f32 acc*scale + aux[m][n]
 */
#define BM 128
#define BN 64
#define BK 32
#define APITCH 40
#define BPITCH 40

struct __align__(16) U128 { unsigned int x[4]; };
union V16U { U128 q[2]; v16bf v; };

__global__ __launch_bounds__(256) void k_gemm_bf16(
    const bf16* __restrict__ A, int lda, long long strA,
    const bf16* __restrict__ B, int ldb, long long strB, int bkmajor,
    int M, int N, int K, int Kvalid,
    const float* __restrict__ bias,
    int mode, float scale,
    const float* __restrict__ aux, int auxld, long long strAux,
    float* __restrict__ outF, bf16* __restrict__ outB, int ldc, long long strOut)
{
  __shared__ __align__(16) bf16 As[2][BM * APITCH];
  __shared__ __align__(16) bf16 Bs[2][BN * BPITCH];

  long long bz = blockIdx.z;
  A += bz * strA;
  B += bz * strB;
  if (aux) aux += bz * strAux;
  if (outF) outF += bz * strOut;
  if (outB) outB += bz * strOut;

  const int tid  = threadIdx.x;
  const int lane = tid & 31;
  const int wave = tid >> 5;
  const int wm   = wave >> 1;   // 0..3  (M direction)
  const int wn   = wave & 1;    // 0..1  (N direction)
  const int l16  = lane & 15;
  const int half = lane >> 4;
  const int m0   = blockIdx.y * BM;
  const int n0   = blockIdx.x * BN;

  v8f acc[2][2] = {};

  // branchless (address-clamped, value-masked) staging; async fast path when
  // the whole tile is in-bounds.
  auto stage = [&](int k0, bf16* Ad, bf16* Bd) {
    const bool fullA = (m0 + BM <= M) && (k0 + BK <= Kvalid);
    #pragma unroll
    for (int ci = 0; ci < 2; ++ci) {
      int linear = tid * 16 + ci * 8;
      int r = linear >> 5, kk = linear & 31;
      int gm = m0 + r, gk = k0 + kk;
      bf16* dst = &Ad[r * APITCH + kk];
      if (HAVE_ASYNC && fullA) {
        async_copy16(A + (size_t)gm * lda + gk, dst);
      } else {
        bool p = (gm < M) && (gk < Kvalid);
        const bf16* src = A + (size_t)(p ? gm : 0) * lda + (p ? gk : 0);
        U128 v = *(const U128*)src;
        unsigned msk = p ? 0xFFFFFFFFu : 0u;
        v.x[0] &= msk; v.x[1] &= msk; v.x[2] &= msk; v.x[3] &= msk;
        *(U128*)dst = v;
      }
    }
    if (!bkmajor) {
      const bool fullB = (n0 + BN <= N) && (k0 + BK <= Kvalid);
      int linear = tid * 8;
      int nn = linear >> 5, kk = linear & 31;
      int gn = n0 + nn, gk = k0 + kk;
      bf16* dst = &Bd[nn * BPITCH + kk];
      if (HAVE_ASYNC && fullB) {
        async_copy16(B + (size_t)gn * ldb + gk, dst);
      } else {
        bool p = (gn < N) && (gk < Kvalid);
        const bf16* src = B + (size_t)(p ? gn : 0) * ldb + (p ? gk : 0);
        U128 v = *(const U128*)src;
        unsigned msk = p ? 0xFFFFFFFFu : 0u;
        v.x[0] &= msk; v.x[1] &= msk; v.x[2] &= msk; v.x[3] &= msk;
        *(U128*)dst = v;
      }
    } else {
      // B stored [K][N]: load 8 contiguous n for one k, scatter-transpose to LDS [n][k]
      int kk = tid >> 3, nn0 = (tid & 7) * 8;
      int gk = k0 + kk, gn0 = n0 + nn0;
      bool p = (gk < Kvalid) && (gn0 < N);
      const bf16* src = B + (size_t)(p ? gk : 0) * ldb + (p ? gn0 : 0);
      U128 v = *(const U128*)src;
      unsigned msk = p ? 0xFFFFFFFFu : 0u;
      v.x[0] &= msk; v.x[1] &= msk; v.x[2] &= msk; v.x[3] &= msk;
      const bf16* pv = (const bf16*)&v;
      #pragma unroll
      for (int e = 0; e < 8; ++e) Bd[(nn0 + e) * BPITCH + kk] = pv[e];
    }
  };

  int buf = 0;
  stage(0, As[0], Bs[0]);
  wait_async();
  __syncthreads();

  for (int k0 = 0; k0 < K; k0 += BK) {
    if (k0 + BK < K) stage(k0 + BK, As[buf ^ 1], Bs[buf ^ 1]);   // prefetch next tile

    const bf16* Ab = As[buf];
    const bf16* Bb = Bs[buf];
    v16bf afrag[2], bfrag[2];
    #pragma unroll
    for (int i = 0; i < 2; ++i) {
      const bf16* pa = &Ab[(wm * 32 + i * 16 + l16) * APITCH + half * 8];
      V16U t;
      t.q[0] = *(const U128*)pa;          // K = kb .. kb+7
      t.q[1] = *(const U128*)(pa + 16);   // K = kb+16 .. kb+23
      afrag[i] = t.v;
    }
    #pragma unroll
    for (int j = 0; j < 2; ++j) {
      const bf16* pb = &Bb[(wn * 32 + j * 16 + l16) * BPITCH + half * 8];
      V16U t;
      t.q[0] = *(const U128*)pb;          // K = kb .. kb+7
      t.q[1] = *(const U128*)(pb + 16);   // K = kb+16 .. kb+23
      bfrag[j] = t.v;
    }
    #pragma unroll
    for (int i = 0; i < 2; ++i)
      #pragma unroll
      for (int j = 0; j < 2; ++j)
        acc[i][j] = __builtin_amdgcn_wmma_f32_16x16x32_bf16(
            false, afrag[i], false, bfrag[j], (short)0, acc[i][j], false, false);

    wait_async();
    __syncthreads();
    buf ^= 1;
  }

  // ---- epilogue (C/D layout: VGPR v -> row v + half*8, col = lane&15)
  #pragma unroll
  for (int i = 0; i < 2; ++i) {
    #pragma unroll
    for (int j = 0; j < 2; ++j) {
      int col = n0 + wn * 32 + j * 16 + l16;
      if (col >= N) continue;
      #pragma unroll
      for (int v = 0; v < 8; ++v) {
        int row = m0 + wm * 32 + i * 16 + half * 8 + v;
        if (row >= M) continue;
        float val = acc[i][j][v];
        if (bias) val += bias[col];
        if (mode == 2) val = val * scale + aux[(size_t)row * auxld + col];
        if (mode == 1) outB[(size_t)row * ldc + col] = (bf16)val;
        else          outF[(size_t)row * ldc + col] = val;
      }
    }
  }
}

/* ---------------- small kernels ---------------- */
__device__ __forceinline__ float sigm(float x) { return 1.0f / (1.0f + __expf(-x)); }

__global__ void k_cvt_bf16(const float* __restrict__ in, bf16* __restrict__ out, int n) {
  int i = (blockIdx.x * 256 + threadIdx.x) * 4;
  if (i + 3 < n) {
    float4 v = *(const float4*)(in + i);
    out[i] = (bf16)v.x; out[i + 1] = (bf16)v.y; out[i + 2] = (bf16)v.z; out[i + 3] = (bf16)v.w;
  } else {
    for (; i < n; ++i) out[i] = (bf16)in[i];
  }
}

__global__ __launch_bounds__(256) void k_layernorm(
    const float* __restrict__ in, const float* __restrict__ w,
    float* __restrict__ outF, bf16* __restrict__ outB, int width)
{
  __shared__ float r1[256], r2[256];
  int row = blockIdx.x, tid = threadIdx.x;
  const float* p = in + (size_t)row * width;
  float s1 = 0.f, s2 = 0.f;
  for (int c = tid; c < width; c += 256) { float v = p[c]; s1 += v; s2 += v * v; }
  r1[tid] = s1; r2[tid] = s2; __syncthreads();
  for (int s = 128; s > 0; s >>= 1) {
    if (tid < s) { r1[tid] += r1[tid + s]; r2[tid] += r2[tid + s]; }
    __syncthreads();
  }
  float mean = r1[0] / width;
  float var  = r2[0] / width - mean * mean;
  float rstd = rsqrtf(var + 1e-5f);
  for (int c = tid; c < width; c += 256) {
    float y = (p[c] - mean) * rstd;
    if (w)    y *= w[c];
    if (outF) outF[(size_t)row * width + c] = y;
    if (outB) outB[(size_t)row * width + c] = (bf16)y;
  }
}

__global__ void k_adaln_combine(const float* __restrict__ pb, const float* __restrict__ an,
                                const float* __restrict__ pnb, bf16* __restrict__ out, int n) {
  int i = blockIdx.x * 256 + threadIdx.x;
  if (i < n) out[i] = (bf16)sigm(pb[i] * an[i] + pnb[i]);
}

/* LN(z) over 64 ch + project to 16 heads; out is (S,S,H) flat.  One wave = one (x,y). */
__global__ __launch_bounds__(256) void k_biasproj(
    const float* __restrict__ z, const float* __restrict__ pw, const float* __restrict__ pbv,
    const float* __restrict__ bw, const float* __restrict__ bb, float* __restrict__ out)
{
  __shared__ float sh[8][64];
  int wave = threadIdx.x >> 5, lane = threadIdx.x & 31;
  size_t p = (size_t)blockIdx.x * 8 + wave;
  float2 v = *(const float2*)(z + p * 64 + lane * 2);
  float s1 = v.x + v.y, s2 = v.x * v.x + v.y * v.y;
  for (int off = 16; off > 0; off >>= 1) {
    s1 += __shfl_xor(s1, off, 32);
    s2 += __shfl_xor(s2, off, 32);
  }
  float mean = s1 * (1.f / 64.f);
  float var  = s2 * (1.f / 64.f) - mean * mean;
  float rstd = rsqrtf(var + 1e-5f);
  sh[wave][lane * 2]     = (v.x - mean) * rstd * pw[lane * 2]     + pbv[lane * 2];
  sh[wave][lane * 2 + 1] = (v.y - mean) * rstd * pw[lane * 2 + 1] + pbv[lane * 2 + 1];
  __syncthreads();
  if (lane < 16) {
    float acc = bb[lane];
    #pragma unroll
    for (int c = 0; c < 64; ++c) acc += sh[wave][c] * bw[lane * 64 + c];
    out[p * 16 + lane] = acc;
  }
}

/* softmax over last dim of scores[h][x][:] then write transposed bf16 At[h][y][x] */
__global__ __launch_bounds__(256) void k_softmax_T(
    const float* __restrict__ scor, bf16* __restrict__ At, int n)
{
  __shared__ float red[256];
  int h = blockIdx.y, x = blockIdx.x, tid = threadIdx.x;
  const float* row = scor + ((size_t)h * n + x) * n;
  bf16* at = At + (size_t)h * n * n;
  float vals[4];
  float m = -3.4e38f;
  #pragma unroll
  for (int j = 0; j < 4; ++j) { vals[j] = row[tid + j * 256]; m = fmaxf(m, vals[j]); }
  red[tid] = m; __syncthreads();
  for (int s = 128; s > 0; s >>= 1) { if (tid < s) red[tid] = fmaxf(red[tid], red[tid + s]); __syncthreads(); }
  m = red[0]; __syncthreads();
  float sum = 0.f;
  #pragma unroll
  for (int j = 0; j < 4; ++j) { vals[j] = __expf(vals[j] - m); sum += vals[j]; }
  red[tid] = sum; __syncthreads();
  for (int s = 128; s > 0; s >>= 1) { if (tid < s) red[tid] += red[tid + s]; __syncthreads(); }
  float inv = 1.f / red[0];
  #pragma unroll
  for (int j = 0; j < 4; ++j) {
    int y = tid + j * 256;
    at[(size_t)y * n + x] = (bf16)(vals[j] * inv);
  }
}

__global__ void k_gate_o(const bf16* __restrict__ kvg, const float* __restrict__ o,
                         bf16* __restrict__ o2, int n) {
  int i = blockIdx.x * 256 + threadIdx.x;
  if (i >= n) return;
  int h = i / (S_ * C_), r = i % (S_ * C_);
  int j = r / C_, c = r % C_;
  float g = (float)kvg[(size_t)h * (S_ * 3 * C_) + (size_t)j * (3 * C_) + 2 * C_ + c];
  o2[i] = (bf16)(sigm(g) * o[i]);
}

__global__ void k_gate_attn(const float* __restrict__ gate, const float* __restrict__ attn1,
                            float* __restrict__ out, int n) {
  int i = blockIdx.x * 256 + threadIdx.x;
  if (i < n) out[i] = sigm(gate[i]) * attn1[i];
}

__global__ void k_swiglu(const float* __restrict__ h12, bf16* __restrict__ bb, int n) {
  int i = blockIdx.x * 256 + threadIdx.x;
  if (i >= n) return;
  int row = i / NCA_, c = i % NCA_;
  float h1 = h12[(size_t)row * (2 * NCA_) + c];
  float h2 = h12[(size_t)row * (2 * NCA_) + NCA_ + c];
  bb[i] = (bf16)(h1 * sigm(h1) * h2);
}

__global__ void k_final(const float* __restrict__ attnb, const float* __restrict__ sg,
                        const float* __restrict__ bbp, float* __restrict__ out, int n) {
  int i = blockIdx.x * 256 + threadIdx.x;
  if (i < n) out[i] = attnb[i] + sigm(sg[i] * bbp[i]);
}

/* ---------------- host ---------------- */
extern "C" void kernel_launch(void* const* d_in, const int* in_sizes, int n_in,
                              void* d_out, int out_size, void* d_ws, size_t ws_size,
                              hipStream_t stream)
{
  (void)in_sizes; (void)n_in; (void)out_size; (void)ws_size;
  const float* a_in0     = (const float*)d_in[0];
  const float* s_in      = (const float*)d_in[1];
  const float* z_in      = (const float*)d_in[2];
  const float* attn_sn_w = (const float*)d_in[3];
  const float* attn_pb_w = (const float*)d_in[4];
  const float* attn_pb_b = (const float*)d_in[5];
  const float* attn_pnb_w= (const float*)d_in[6];
  const float* pair_w    = (const float*)d_in[7];
  const float* pair_b    = (const float*)d_in[8];
  const float* q_w       = (const float*)d_in[9];
  const float* q_b       = (const float*)d_in[10];
  const float* kvg_w     = (const float*)d_in[11];
  const float* bias_w    = (const float*)d_in[12];
  const float* bias_b    = (const float*)d_in[13];
  const float* ao_w      = (const float*)d_in[14];
  const float* out_w     = (const float*)d_in[15];
  const float* out_b     = (const float*)d_in[16];
  const float* tr_sn_w   = (const float*)d_in[17];
  const float* tr_pb_w   = (const float*)d_in[18];
  const float* tr_pb_b   = (const float*)d_in[19];
  const float* tr_pnb_w  = (const float*)d_in[20];
  const float* tr_a_w    = (const float*)d_in[21];
  const float* tr_s_w    = (const float*)d_in[22];
  const float* tr_s_b    = (const float*)d_in[23];
  const float* tr_b_w    = (const float*)d_in[24];

  size_t off = 0;
  auto alloc = [&](size_t bytes) -> void* {
    void* p = (char*)d_ws + off;
    off += (bytes + 255) & ~(size_t)255;
    return p;
  };
  auto cvt = [&](const float* src, size_t n) -> bf16* {
    bf16* dst = (bf16*)alloc(n * sizeof(bf16));
    k_cvt_bf16<<<dim3((unsigned)((n + 1023) / 1024)), 256, 0, stream>>>(src, dst, (int)n);
    return dst;
  };

  // bf16 weights (all layers at once)
  bf16* wpb   = cvt(attn_pb_w, (size_t)L_ * CA_ * CS_);
  bf16* wpnb  = cvt(attn_pnb_w,(size_t)L_ * CA_ * CS_);
  bf16* wq    = cvt(q_w,       (size_t)L_ * CA_ * CA_);
  bf16* wkvg  = cvt(kvg_w,     (size_t)L_ * KVG_ * CA_);
  bf16* wao   = cvt(ao_w,      (size_t)L_ * CA_ * CA_);
  bf16* wout  = cvt(out_w,     (size_t)L_ * CA_ * CS_);
  bf16* wtpb  = cvt(tr_pb_w,   (size_t)L_ * CA_ * CS_);
  bf16* wtpnb = cvt(tr_pnb_w,  (size_t)L_ * CA_ * CS_);
  bf16* wtra  = cvt(tr_a_w,    (size_t)L_ * 2 * NCA_ * CA_);
  bf16* wtrs  = cvt(tr_s_w,    (size_t)L_ * CA_ * CS_);
  bf16* wtrb  = cvt(tr_b_w,    (size_t)L_ * CA_ * NCA_);
  bf16* sbuf  = cvt(s_in,      (size_t)S_ * CS_);

  float* an    = (float*)alloc((size_t)S_ * CA_ * 4);
  float* tmp1  = (float*)alloc((size_t)S_ * CA_ * 4);
  float* tmp2  = (float*)alloc((size_t)S_ * CA_ * 4);
  bf16*  snb   = (bf16*) alloc((size_t)S_ * CS_ * 2);
  bf16*  a2b   = (bf16*) alloc((size_t)S_ * CA_ * 2);
  bf16*  qb    = (bf16*) alloc((size_t)S_ * CA_ * 2);
  bf16*  kvgb  = (bf16*) alloc((size_t)S_ * KVG_ * 2);
  float* bproj = (float*)alloc((size_t)S_ * S_ * H_ * 4);
  float* scor  = (float*)alloc((size_t)H_ * S_ * S_ * 4);
  bf16*  At    = (bf16*) alloc((size_t)H_ * S_ * S_ * 2);
  float* ofl   = (float*)alloc((size_t)S_ * CA_ * 4);
  bf16*  o2b   = (bf16*) alloc((size_t)S_ * CA_ * 2);
  float* attn1 = (float*)alloc((size_t)S_ * CA_ * 4);
  float* gateb = (float*)alloc((size_t)S_ * CA_ * 4);
  float* attnb = (float*)alloc((size_t)S_ * CA_ * 4);
  float* h12   = (float*)alloc((size_t)S_ * 2 * NCA_ * 4);
  bf16*  bbb   = (bf16*) alloc((size_t)S_ * NCA_ * 2);
  float* bbp   = (float*)alloc((size_t)S_ * CA_ * 4);
  float* sg    = (float*)alloc((size_t)S_ * CA_ * 4);
  float* aping = (float*)alloc((size_t)S_ * CA_ * 4);

  auto gemm = [&](const bf16* Ap, int lda, long long sA,
                  const bf16* Bp, int ldb, long long sB, int bkm,
                  int M, int N, int Kvalid, const float* bias,
                  int mode, float scale, const float* aux, int auxld, long long sAux,
                  float* oF, bf16* oB, int ldc, long long sOut, int batch) {
    int K = (Kvalid + BK - 1) / BK * BK;
    dim3 g((N + BN - 1) / BN, (M + BM - 1) / BM, batch);
    k_gemm_bf16<<<g, 256, 0, stream>>>(Ap, lda, sA, Bp, ldb, sB, bkm, M, N, K, Kvalid,
                                       bias, mode, scale, aux, auxld, sAux, oF, oB, ldc, sOut);
  };
  auto ew = [](int n) { return dim3((unsigned)((n + 255) / 256)); };

  for (int l = 0; l < L_; ++l) {
    const float* a_in  = (l == 0) ? a_in0 : aping;
    float*       a_out = (l == L_ - 1) ? (float*)d_out : aping;
    size_t lCS  = (size_t)l * CA_ * CS_;
    size_t lCA2 = (size_t)l * CA_ * CA_;

    // ---- AttentionPairBias
    k_layernorm<<<S_, 256, 0, stream>>>(a_in, nullptr, an, nullptr, CA_);
    k_layernorm<<<S_, 256, 0, stream>>>(s_in, attn_sn_w + (size_t)l * CS_, nullptr, snb, CS_);
    gemm(snb, CS_, 0, wpb + lCS, CS_, 0, 0, S_, CA_, CS_, attn_pb_b + (size_t)l * CA_,
         0, 1.f, nullptr, 0, 0, tmp1, nullptr, CA_, 0, 1);
    gemm(snb, CS_, 0, wpnb + lCS, CS_, 0, 0, S_, CA_, CS_, nullptr,
         0, 1.f, nullptr, 0, 0, tmp2, nullptr, CA_, 0, 1);
    k_adaln_combine<<<ew(S_ * CA_), 256, 0, stream>>>(tmp1, an, tmp2, a2b, S_ * CA_);
    gemm(a2b, CA_, 0, wq + lCA2, CA_, 0, 0, S_, CA_, CA_, q_b + (size_t)l * CA_,
         1, 1.f, nullptr, 0, 0, nullptr, qb, CA_, 0, 1);
    gemm(a2b, CA_, 0, wkvg + (size_t)l * KVG_ * CA_, CA_, 0, 0, S_, KVG_, CA_, nullptr,
         1, 1.f, nullptr, 0, 0, nullptr, kvgb, KVG_, 0, 1);
    k_biasproj<<<(S_ * S_) / 8, 256, 0, stream>>>(
        z_in, pair_w + (size_t)l * CZ_, pair_b + (size_t)l * CZ_,
        bias_w + (size_t)l * H_ * CZ_, bias_b + (size_t)l * H_, bproj);
    // scores[x,y] = k[x]·q[y]/C + bias[x,y]   (batched over heads)
    gemm(kvgb, 3 * C_, (long long)S_ * 3 * C_,
         qb, C_, (long long)S_ * C_, 0,
         S_, S_, C_, nullptr, 2, 1.0f / (float)C_, bproj, S_, (long long)S_ * S_,
         scor, nullptr, S_, (long long)S_ * S_, H_);
    k_softmax_T<<<dim3(S_, H_), 256, 0, stream>>>(scor, At, S_);
    // o[j,c] = sum_x At[j,x] * v[x,c]
    gemm(At, S_, (long long)S_ * S_,
         kvgb + C_, 3 * C_, (long long)S_ * 3 * C_, 1,
         S_, C_, S_, nullptr, 0, 1.f, nullptr, 0, 0,
         ofl, nullptr, C_, (long long)S_ * C_, H_);
    k_gate_o<<<ew(S_ * CA_), 256, 0, stream>>>(kvgb, ofl, o2b, S_ * CA_);
    gemm(o2b, CA_, 0, wao + lCA2, CA_, 0, 0, S_, CA_, CA_, nullptr,
         0, 1.f, nullptr, 0, 0, attn1, nullptr, CA_, 0, 1);
    gemm(sbuf, CS_, 0, wout + lCS, CS_, 0, 0, S_, CA_, CS_, out_b + (size_t)l * CA_,
         0, 1.f, nullptr, 0, 0, gateb, nullptr, CA_, 0, 1);
    k_gate_attn<<<ew(S_ * CA_), 256, 0, stream>>>(gateb, attn1, attnb, S_ * CA_);

    // ---- ConditionedTransitionBlock
    k_layernorm<<<S_, 256, 0, stream>>>(s_in, tr_sn_w + (size_t)l * CS_, nullptr, snb, CS_);
    gemm(snb, CS_, 0, wtpb + lCS, CS_, 0, 0, S_, CA_, CS_, tr_pb_b + (size_t)l * CA_,
         0, 1.f, nullptr, 0, 0, tmp1, nullptr, CA_, 0, 1);
    gemm(snb, CS_, 0, wtpnb + lCS, CS_, 0, 0, S_, CA_, CS_, nullptr,
         0, 1.f, nullptr, 0, 0, tmp2, nullptr, CA_, 0, 1);
    k_adaln_combine<<<ew(S_ * CA_), 256, 0, stream>>>(tmp1, an, tmp2, a2b, S_ * CA_);
    gemm(a2b, CA_, 0, wtra + (size_t)l * 2 * NCA_ * CA_, CA_, 0, 0, S_, 2 * NCA_, CA_, nullptr,
         0, 1.f, nullptr, 0, 0, h12, nullptr, 2 * NCA_, 0, 1);
    k_swiglu<<<ew(S_ * NCA_), 256, 0, stream>>>(h12, bbb, S_ * NCA_);
    gemm(bbb, NCA_, 0, wtrb + (size_t)l * CA_ * NCA_, NCA_, 0, 0, S_, CA_, NCA_, nullptr,
         0, 1.f, nullptr, 0, 0, bbp, nullptr, CA_, 0, 1);
    gemm(sbuf, CS_, 0, wtrs + lCS, CS_, 0, 0, S_, CA_, CS_, tr_s_b + (size_t)l * CA_,
         0, 1.f, nullptr, 0, 0, sg, nullptr, CA_, 0, 1);
    k_final<<<ew(S_ * CA_), 256, 0, stream>>>(attnb, sg, bbp, a_out, S_ * CA_);
  }
}
